// MixMLP_18030272708891
// MI455X (gfx1250) — matrix operand order repaired and checked by reading, not compiled, not run on hardware
//
#include <hip/hip_runtime.h>
#include <hip/hip_bf16.h>
#include <math.h>

typedef __attribute__((ext_vector_type(16))) __bf16 v16bf;
typedef __attribute__((ext_vector_type(8)))  float  v8f;

#define NNODES 256
#define BATCH  512
#define NE     32640          // edges = 256*255/2
#define OUTC   65280          // 2*NE
#define KC     2048           // concat K (1024 big + 1024 small)
#define KB     32             // K per WMMA step
#define NKB    64             // KC/KB

// ---------- helpers ----------
__device__ __forceinline__ unsigned short f2bf(float f) {
    unsigned int u = __float_as_uint(f);
    u += 0x7FFFu + ((u >> 16) & 1u);   // round to nearest even
    return (unsigned short)(u >> 16);
}

__device__ __forceinline__ int edge_offs(int i) {   // #edges before row i
    return 255 * i - (i * (i - 1)) / 2;
}

// ---------- edge (i,j) table: e -> (i<<8)|j ----------
__global__ void edge_table_k(unsigned int* __restrict__ tbl) {
    int e = blockIdx.x * 256 + threadIdx.x;
    if (e >= NE) return;
    float d = 511.0f * 511.0f - 8.0f * (float)e;
    int i = (int)floorf((511.0f - sqrtf(d)) * 0.5f);
    if (i < 0) i = 0; if (i > 254) i = 254;
    while (i > 0 && edge_offs(i) > e) --i;
    while (i < 254 && edge_offs(i + 1) <= e) ++i;
    int j = e - edge_offs(i) + i + 1;
    tbl[e] = ((unsigned)i << 8) | (unsigned)j;
}

// ---------- zero the diagonal (scatter covers everything else) ----------
__global__ void zero_diag_k(float* __restrict__ out) {
    int idx = blockIdx.x * 256 + threadIdx.x;   // 512*256 threads
    int b = idx >> 8, i = idx & 255;
    out[(size_t)b * 65536 + (size_t)i * 257] = 0.0f;
}

// ---------- small fp32 GEMM + bias + relu (layers 0..2) ----------
__global__ __launch_bounds__(256)
void mlp_layer_k(const float* __restrict__ in, const float* __restrict__ w,
                 const float* __restrict__ bias, float* __restrict__ out,
                 int K, int N) {
    __shared__ float sx[64];
    const int n = blockIdx.x * 256 + threadIdx.x;
    const int b = blockIdx.y;
    float acc = bias[n];
    for (int k0 = 0; k0 < K; k0 += 64) {
        __syncthreads();
        if (threadIdx.x < 64) sx[threadIdx.x] = in[(size_t)b * K + k0 + threadIdx.x];
        __syncthreads();
        #pragma unroll 16
        for (int kk = 0; kk < 64; ++kk)
            acc = fmaf(sx[kk], w[(size_t)(k0 + kk) * N + n], acc);
    }
    out[(size_t)b * N + n] = fmaxf(acc, 0.0f);
}

// ---------- pack H = [m*h_big , (1-m)*h_small] into bf16 A-fragment layout ----------
// A 16x32 bf16 fragment (ISA 7.12.2): lanes 0-15: M=0..15, ushort t=0..7 -> K=0..7,
// t=8..15 -> K=16..23 ; lanes 16-31: K=8..15 / 24..31.
__global__ void pack_h_k(const float* __restrict__ x, const float* __restrict__ hb2,
                         const float* __restrict__ hs2, unsigned short* __restrict__ Hp) {
    int idx = blockIdx.x * 256 + threadIdx.x;   // 512*2048
    int b = idx >> 11;
    int k = idx & 2047;
    bool mbig = x[(size_t)b * 64] > 0.0f;
    float v;
    if (k < 1024) v = mbig ? hb2[(size_t)b * 1024 + k] : 0.0f;
    else          v = mbig ? 0.0f : hs2[(size_t)b * 1024 + (k - 1024)];
    int mb = b >> 4, kb = k >> 5, kk = k & 31;
    int lane = (b & 15) + 16 * ((kk >> 3) & 1);
    int tt   = (kk & 7) + ((kk & 16) ? 8 : 0);
    Hp[(size_t)(mb * NKB + kb) * 512 + lane * 16 + tt] = f2bf(v);
}

// ---------- main edge GEMM + fused gumbel-argmax + symmetric scatter ----------
// Block: 64 rows x 32 edges, 8 waves (4 M-subtiles x 2 edge-subtiles).
// Each wave: two v8f accumulators (channel 0 / channel 1 of each edge).
__global__ __launch_bounds__(256)
void edge_gemm_k(const unsigned short* __restrict__ Hp,
                 const float* __restrict__ bw3, const float* __restrict__ sw3,
                 const float* __restrict__ bb3, const float* __restrict__ sb3,
                 const float* __restrict__ x,   const float* __restrict__ gumbel,
                 const unsigned int* __restrict__ tbl, float* __restrict__ out) {
    // B 32x16 bf16 fragment: lanes 0-15 hold K=0..7 (t=0..7) / 16..23 (t=8..15), lanes 16-31 K=8..15 / 24..31.
    __shared__ unsigned short ldsB[2][2][32][16];   // [ch][edge-subtile][lane][t]

    const int tid  = threadIdx.x;
    const int lane = tid & 31;
    const int wave = tid >> 5;
    const int wm   = wave & 3;            // M subtile 0..3
    const int wn   = wave >> 2;           // edge subtile 0..1
    const int mb    = blockIdx.y * 4 + wm;       // 16-row block index
    const int mBase = mb * 16;
    const int eBase = blockIdx.x * 32 + wn * 16;
    const int colBase = blockIdx.x * 64;         // fp32 column base in w3

    v8f c0 = {};
    v8f c1 = {};

    // cooperative staging: thread -> (row kk, 8 consecutive cols at cc0)
    const int kk  = tid >> 3;        // 0..31
    const int cc0 = (tid & 7) * 8;   // 0..56
    const int halfB = (kk >> 3) & 1; // lane half by K
    const int ttB   = (kk & 7) + ((kk & 16) ? 8 : 0);

    for (int kb = 0; kb < NKB; ++kb) {
        const int kg = kb * KB + kk;
        const float* wsrc = (kg < 1024)
            ? (bw3 + (size_t)kg * OUTC)
            : (sw3 + (size_t)(kg - 1024) * OUTC);
        const float4 va = *(const float4*)(wsrc + colBase + cc0);
        const float4 vb = *(const float4*)(wsrc + colBase + cc0 + 4);
        if (kb + 1 < NKB) {
            const int kg2 = kg + KB;
            const float* wnext = (kg2 < 1024)
                ? (bw3 + (size_t)kg2 * OUTC)
                : (sw3 + (size_t)(kg2 - 1024) * OUTC);
            __builtin_prefetch(wnext + colBase + cc0, 0, 1);
        }
        float vals[8] = {va.x, va.y, va.z, va.w, vb.x, vb.y, vb.z, vb.w};
        #pragma unroll
        for (int q = 0; q < 8; ++q) {
            int cc = cc0 + q;
            int ch = cc & 1;          // even col = channel 0, odd = channel 1
            int el = cc >> 1;         // local edge 0..31
            ldsB[ch][el >> 4][(el & 15) + 16 * halfB][ttB] = f2bf(vals[q]);
        }
        __syncthreads();

        v16bf a  = *(const v16bf*)(Hp + ((size_t)(mb * NKB + kb) * 512 + lane * 16));
        v16bf b0 = *(const v16bf*)(&ldsB[0][wn][lane][0]);
        v16bf b1 = *(const v16bf*)(&ldsB[1][wn][lane][0]);
        c0 = __builtin_amdgcn_wmma_f32_16x16x32_bf16(false, a, false, b0,
                                                     (short)0, c0, false, false);
        c1 = __builtin_amdgcn_wmma_f32_16x16x32_bf16(false, a, false, b1,
                                                     (short)0, c1, false, false);
        __syncthreads();
    }

    // epilogue: C 16x16 f32 layout -> lane 0-15: N=lane, M=r ; lane 16-31: N=lane-16, M=r+8
    const int nloc  = lane & 15;
    const int halfM = lane >> 4;
    const int e  = eBase + nloc;
    const unsigned ij = tbl[e];
    const int iu = (int)(ij >> 8), ju = (int)(ij & 255);
    const float bb0v = bb3[2 * e], bb1v = bb3[2 * e + 1];
    const float sb0v = sb3[2 * e], sb1v = sb3[2 * e + 1];
    #pragma unroll
    for (int r = 0; r < 8; ++r) {
        int m = mBase + r + halfM * 8;
        bool mbig = x[(size_t)m * 64] > 0.0f;
        size_t gidx = ((size_t)m * NE + e) * 2;
        float g0 = c0[r] + (mbig ? bb0v : sb0v) + gumbel[gidx];
        float g1 = c1[r] + (mbig ? bb1v : sb1v) + gumbel[gidx + 1];
        float val = (g0 >= g1) ? 1.0f : 0.0f;   // argmax tie -> index 0 -> 1.0
        size_t ob = (size_t)m * 65536;
        out[ob + (size_t)iu * 256 + ju] = val;
        out[ob + (size_t)ju * 256 + iu] = val;
    }
}

// ---------- launcher ----------
extern "C" void kernel_launch(void* const* d_in, const int* in_sizes, int n_in,
                              void* d_out, int out_size, void* d_ws, size_t ws_size,
                              hipStream_t stream) {
    (void)in_sizes; (void)n_in; (void)out_size; (void)ws_size;
    const float* x      = (const float*)d_in[0];
    const float* gumbel = (const float*)d_in[1];
    const float* bw0 = (const float*)d_in[2],  *bb0 = (const float*)d_in[3];
    const float* sw0 = (const float*)d_in[4],  *sb0 = (const float*)d_in[5];
    const float* bw1 = (const float*)d_in[6],  *bb1 = (const float*)d_in[7];
    const float* sw1 = (const float*)d_in[8],  *sb1 = (const float*)d_in[9];
    const float* bw2 = (const float*)d_in[10], *bb2 = (const float*)d_in[11];
    const float* sw2 = (const float*)d_in[12], *sb2 = (const float*)d_in[13];
    const float* bw3 = (const float*)d_in[14], *bb3 = (const float*)d_in[15];
    const float* sw3 = (const float*)d_in[16], *sb3 = (const float*)d_in[17];
    float* out = (float*)d_out;

    char* ws = (char*)d_ws;                       // ~9.6 MB used
    float* hb0 = (float*)(ws + 0);                // 512*256  f32
    float* hb1 = (float*)(ws + 524288);           // 512*512  f32
    float* hb2 = (float*)(ws + 1572864);          // 512*1024 f32
    float* hs0 = (float*)(ws + 3670016);
    float* hs1 = (float*)(ws + 4194304);
    float* hs2 = (float*)(ws + 5242880);
    unsigned short* Hp = (unsigned short*)(ws + 7340032);   // 512*2048 bf16 swizzled
    unsigned int*  tbl = (unsigned int*)(ws + 9437184);     // 32640 u32

    edge_table_k<<<(NE + 255) / 256, 256, 0, stream>>>(tbl);
    zero_diag_k<<<512, 256, 0, stream>>>(out);

    mlp_layer_k<<<dim3(1, BATCH), 256, 0, stream>>>(x,   bw0, bb0, hb0,  64,  256);
    mlp_layer_k<<<dim3(2, BATCH), 256, 0, stream>>>(hb0, bw1, bb1, hb1, 256,  512);
    mlp_layer_k<<<dim3(4, BATCH), 256, 0, stream>>>(hb1, bw2, bb2, hb2, 512, 1024);
    mlp_layer_k<<<dim3(1, BATCH), 256, 0, stream>>>(x,   sw0, sb0, hs0,  64,  256);
    mlp_layer_k<<<dim3(2, BATCH), 256, 0, stream>>>(hs0, sw1, sb1, hs1, 256,  512);
    mlp_layer_k<<<dim3(4, BATCH), 256, 0, stream>>>(hs1, sw2, sb2, hs2, 512, 1024);

    pack_h_k<<<(BATCH * KC) / 256, 256, 0, stream>>>(x, hb2, hs2, Hp);

    edge_gemm_k<<<dim3(NE / 32, BATCH / 64), 256, 0, stream>>>(
        Hp, bw3, sw3, bb3, sb3, x, gumbel, tbl, out);
}
